// AutoformerForecast_22007412424961
// MI455X (gfx1250) — matrix operand order repaired and checked by reading, not compiled, not run on hardware
//
#include <hip/hip_runtime.h>
#include <hip/hip_bf16.h>
#include <math.h>

// ---------------------------------------------------------------------------
// AutoformerForecast for MI455X (gfx1250, wave32).
// B=64, Cin=8, T=2048, D=64, H=24, F=5.  All data fp32 (f16 only inside WMMA).
// ---------------------------------------------------------------------------

#define BATCH   64
#define CIN     8
#define T_LEN   2048
#define LOG_T   11
#define DMODEL  64
#define HORIZON 24
#define NFACT   5
#define LTOT    (T_LEN + HORIZON)   // 2072
#define TOPK    3

typedef _Float16 v16h __attribute__((ext_vector_type(16)));
typedef float    v8f  __attribute__((ext_vector_type(8)));

// ---------------------------------------------------------------------------
// K1: input projection  xe[b,d,t] = sum_c Wi[d,c]*x[b,c,t] + bi[d]
// grid (T/256, B), block 256
// ---------------------------------------------------------------------------
__global__ __launch_bounds__(256) void k_input_proj(
    const float* __restrict__ x, const float* __restrict__ Wi,
    const float* __restrict__ bi, float* __restrict__ xe)
{
    const int b   = blockIdx.y;
    const int t0  = blockIdx.x * 256;
    const int tid = threadIdx.x;

    __shared__ float xs[CIN][256];
    __shared__ float Ws[DMODEL * CIN];
    __shared__ float bs[DMODEL];

    if (tid < DMODEL) bs[tid] = bi[tid];
    for (int i = tid; i < DMODEL * CIN; i += 256) Ws[i] = Wi[i];
    #pragma unroll
    for (int c = 0; c < CIN; ++c)
        xs[c][tid] = x[((size_t)(b * CIN + c)) * T_LEN + t0 + tid];
    __syncthreads();

    for (int d = 0; d < DMODEL; ++d) {
        float a = bs[d];
        #pragma unroll
        for (int c = 0; c < CIN; ++c) a += Ws[d * CIN + c] * xs[c][tid];
        xe[((size_t)(b * DMODEL + d)) * T_LEN + t0 + tid] = a;
    }
}

// ---------------------------------------------------------------------------
// Radix-2 DIT FFT, 2048 points, data in LDS, 256 threads. sign=-1 fwd, +1 inv.
// ---------------------------------------------------------------------------
__device__ inline void fft2048(float* re, float* im, int tid, float sign)
{
    // bit-reversal permutation (each unordered pair swapped exactly once)
    #pragma unroll
    for (int i = 0; i < 8; ++i) {
        int t = tid + 256 * i;
        int j = (int)(__brev((unsigned)t) >> (32 - LOG_T));
        if (j > t) {
            float a = re[t]; re[t] = re[j]; re[j] = a;
            float c = im[t]; im[t] = im[j]; im[j] = c;
        }
    }
    __syncthreads();

    for (int s = 0; s < LOG_T; ++s) {
        const int half = 1 << s;
        #pragma unroll
        for (int i = 0; i < 4; ++i) {
            int idx  = tid + 256 * i;                 // 0..1023 butterflies
            int j    = idx & (half - 1);
            int base = ((idx >> s) << (s + 1)) + j;
            float ang = sign * 6.28318530717958647f * (float)j / (float)(2 << s);
            float wr = __cosf(ang), wi = __sinf(ang);
            float xr = re[base + half], xi = im[base + half];
            float tr = wr * xr - wi * xi;
            float ti = wr * xi + wi * xr;
            float ur = re[base], ui = im[base];
            re[base + half] = ur - tr; im[base + half] = ui - ti;
            re[base]        = ur + tr; im[base]        = ui + ti;
        }
        __syncthreads();
    }
}

// ---------------------------------------------------------------------------
// K2/K4: fused autoformer core per row (b*D+d):
//   trend = AvgPool1d(k=25, count_include_pad);  d = x - trend
//   ac    = circular autocorrelation of d via FFT -> |F|^2 -> inv FFT
//   lags  = top-3 of ac;  seas[t] = mean_k d[t-lag_k] (0 if t<lag)
// grid 4096 blocks, block 256.  LDS ~30.7 KB.
// ---------------------------------------------------------------------------
__global__ __launch_bounds__(256) void k_autoformer_core(
    const float* __restrict__ in, float* __restrict__ trend_out,
    float* __restrict__ seas_out)
{
    const int row = blockIdx.x;
    const int tid = threadIdx.x;
    const float* xrow = in + (size_t)row * T_LEN;
    float* trow = trend_out + (size_t)row * T_LEN;
    float* srow = seas_out  + (size_t)row * T_LEN;

    __shared__ float re[T_LEN];
    __shared__ float im[T_LEN];
    __shared__ float dc[T_LEN];
    __shared__ float cv[256 * TOPK];
    __shared__ int   ci[256 * TOPK];
    __shared__ int   lag3[TOPK];

    #pragma unroll
    for (int i = 0; i < 8; ++i) { int t = tid + 256 * i; re[t] = xrow[t]; }
    __syncthreads();

    // 25-tap centered average (zero pad, always /25), detrend into registers
    float dloc[8], tloc[8];
    #pragma unroll
    for (int i = 0; i < 8; ++i) {
        int t  = tid + 256 * i;
        int lo = (t - 12 < 0) ? 0 : t - 12;
        int hi = (t + 12 > T_LEN - 1) ? (T_LEN - 1) : t + 12;
        float s = 0.f;
        for (int j = lo; j <= hi; ++j) s += re[j];
        tloc[i] = s * (1.0f / 25.0f);
        dloc[i] = re[t] - tloc[i];
    }
    __syncthreads();   // everyone done reading re as raw input
    #pragma unroll
    for (int i = 0; i < 8; ++i) {
        int t = tid + 256 * i;
        trow[t] = tloc[i];
        dc[t]   = dloc[i];
        re[t]   = dloc[i];
        im[t]   = 0.f;
    }
    __syncthreads();

    fft2048(re, im, tid, -1.0f);                      // forward FFT of d

    #pragma unroll
    for (int i = 0; i < 8; ++i) {                     // power spectrum
        int t = tid + 256 * i;
        float a = re[t], c = im[t];
        re[t] = a * a + c * c;
        im[t] = 0.f;
    }
    __syncthreads();

    fft2048(re, im, tid, +1.0f);                      // inverse -> T * ac[t]

    // local top-3 (contiguous chunk of 8 per thread)
    float b0 = -3.0e38f, b1 = -3.0e38f, b2 = -3.0e38f;
    int   i0 = 0, i1 = 0, i2 = 0;
    #pragma unroll
    for (int i = 0; i < 8; ++i) {
        int t = tid * 8 + i;
        float v = re[t];
        if (v > b0)      { b2=b1; i2=i1; b1=b0; i1=i0; b0=v; i0=t; }
        else if (v > b1) { b2=b1; i2=i1; b1=v; i1=t; }
        else if (v > b2) { b2=v; i2=t; }
    }
    cv[tid*3+0]=b0; ci[tid*3+0]=i0;
    cv[tid*3+1]=b1; ci[tid*3+1]=i1;
    cv[tid*3+2]=b2; ci[tid*3+2]=i2;
    __syncthreads();

    if (tid == 0) {
        float g0=-3.0e38f, g1=-3.0e38f, g2=-3.0e38f; int j0=0, j1=0, j2=0;
        for (int c = 0; c < 256 * TOPK; ++c) {
            float v = cv[c]; int t = ci[c];
            if (v > g0)      { g2=g1; j2=j1; g1=g0; j1=j0; g0=v; j0=t; }
            else if (v > g1) { g2=g1; j2=j1; g1=v; j1=t; }
            else if (v > g2) { g2=v; j2=t; }
        }
        lag3[0]=j0; lag3[1]=j1; lag3[2]=j2;
    }
    __syncthreads();

    const int l0 = lag3[0], l1 = lag3[1], l2 = lag3[2];
    #pragma unroll
    for (int i = 0; i < 8; ++i) {
        int t = tid + 256 * i;
        float s = 0.f;
        if (t >= l0) s += dc[t - l0];
        if (t >= l1) s += dc[t - l1];
        if (t >= l2) s += dc[t - l2];
        srow[t] = s * (1.0f / 3.0f);
    }
}

// ---------------------------------------------------------------------------
// K3: WMMA conv1x1 + residual:  u2 = Wp1 @ (seas1 + tr1) + bp1 + tr1
// M=64 (rows of Wp1), K=64 (channels), N=T per batch.
// grid (T/128, B), block 256 = 8 waves.
// Activation tile [64ch x 128t] staged once per block into LDS as f16
// (coalesced loads); A fragments (Wp1) held in VGPRs across the N loop.
// Each wave: 4 N-tiles x 2 K-steps = 8 v_wmma.
// ---------------------------------------------------------------------------
#define NCHUNK  128
#define BSTRIDE 136   // LDS tile row stride in halves: 272B = 68 banks (no 2-way conflict)

__global__ __launch_bounds__(256) void k_conv_wmma(
    const float* __restrict__ seas, const float* __restrict__ tr,
    const float* __restrict__ W, const float* __restrict__ bias,
    float* __restrict__ out)
{
    const int b     = blockIdx.y;
    const int n0    = blockIdx.x * NCHUNK;
    const int tid   = threadIdx.x;
    const int wave  = tid >> 5;             // 0..7
    const int lane  = tid & 31;
    const int mtile = wave & 3;             // 4 tiles cover M=64
    const int ngrp  = wave >> 2;            // 2 groups of 64 columns
    const int lo    = lane & 15;
    const int hi    = lane >> 4;            // lane group 0/1

    __shared__ _Float16 inT[DMODEL * BSTRIDE];   // ~17 KB

    // cooperative, coalesced staging of (seas+tr) -> f16 LDS tile
    for (int idx = tid; idx < DMODEL * NCHUNK; idx += 256) {
        int ch = idx >> 7;              // idx / 128
        int t  = idx & (NCHUNK - 1);
        size_t g = ((size_t)(b * DMODEL + ch)) * T_LEN + n0 + t;
        inT[ch * BSTRIDE + t] = (_Float16)(seas[g] + tr[g]);
    }

    // A fragments: this wave's 16 output channels of Wp1, kept in VGPRs.
    // A (16x32 f16) per-lane K packing: e<8 -> hi*8+e ; e>=8 -> 16+hi*8+(e-8)
    const int rowA = mtile * 16 + lo;
    v16h afrag[2];
    #pragma unroll
    for (int kk = 0; kk < 2; ++kk) {
        #pragma unroll
        for (int e = 0; e < 16; ++e) {
            int ka = kk * 32 + ((e < 8) ? (hi * 8 + e) : (16 + hi * 8 + (e - 8)));
            afrag[kk][e] = (_Float16)W[rowA * DMODEL + ka];
        }
    }
    __syncthreads();

    #pragma unroll
    for (int nt = 0; nt < 4; ++nt) {
        const int cloc = ngrp * 64 + nt * 16 + lo;   // column within the tile
        v8f acc = {};
        #pragma unroll
        for (int kk = 0; kk < 2; ++kk) {
            v16h bmat;
            // B (32x16 f16) per-lane K packing: e<8 -> hi*8+e ; e>=8 -> 16+hi*8+(e-8)
            #pragma unroll
            for (int e = 0; e < 16; ++e)
                bmat[e] = inT[(kk * 32 + ((e < 8) ? (hi * 8 + e) : (16 + hi * 8 + (e - 8)))) * BSTRIDE + cloc];
            acc = __builtin_amdgcn_wmma_f32_16x16x32_f16(
                /*neg_a=*/false, afrag[kk], /*neg_b=*/false, bmat,
                /*c_mod=*/(short)0, acc, /*reuse_a=*/false, /*reuse_b=*/false);
        }
        // C/D layout: VGPR r, lane group hi -> M = mtile*16 + 8*hi + r
        #pragma unroll
        for (int r = 0; r < 8; ++r) {
            int m = mtile * 16 + 8 * hi + r;
            size_t g = ((size_t)(b * DMODEL + m)) * T_LEN + n0 + cloc;
            out[g] = acc[r] + bias[m] + tr[g];
        }
    }
}

// ---------------------------------------------------------------------------
// K5: decoders on last timestep only (exact: 1x1 convs are pointwise in t)
//   h = relu(W1 @ last + b1); out = W2 @ h + b2 ; yhat[b,h,d] = trend+seas
// grid B, block 256
// ---------------------------------------------------------------------------
__global__ __launch_bounds__(256) void k_decoder(
    const float* __restrict__ tr2, const float* __restrict__ seas2,
    const float* __restrict__ Wt1, const float* __restrict__ bt1,
    const float* __restrict__ Wt2, const float* __restrict__ bt2,
    const float* __restrict__ Ws1, const float* __restrict__ bs1,
    const float* __restrict__ Ws2, const float* __restrict__ bs2,
    float* __restrict__ yhat)
{
    const int b = blockIdx.x, tid = threadIdx.x;
    __shared__ float tl[DMODEL], sl[DMODEL], h1[DMODEL], h2[DMODEL];

    if (tid < DMODEL) {
        tl[tid] = tr2 [((size_t)(b * DMODEL + tid)) * T_LEN + (T_LEN - 1)];
        sl[tid] = seas2[((size_t)(b * DMODEL + tid)) * T_LEN + (T_LEN - 1)];
    }
    __syncthreads();
    if (tid < DMODEL) {
        float a1 = bt1[tid], a2 = bs1[tid];
        for (int i = 0; i < DMODEL; ++i) {
            a1 += Wt1[tid * DMODEL + i] * tl[i];
            a2 += Ws1[tid * DMODEL + i] * sl[i];
        }
        h1[tid] = fmaxf(a1, 0.f);
        h2[tid] = fmaxf(a2, 0.f);
    }
    __syncthreads();
    for (int j = tid; j < DMODEL * HORIZON; j += 256) {
        float a = bt2[j] + bs2[j];
        for (int i = 0; i < DMODEL; ++i)
            a += Wt2[j * DMODEL + i] * h1[i] + Ws2[j * DMODEL + i] * h2[i];
        int d = j / HORIZON, h = j % HORIZON;      // flat = d*H + h
        yhat[((size_t)(b * HORIZON + h)) * DMODEL + d] = a;   // [B,H,D]
    }
}

// ---------------------------------------------------------------------------
// Block reductions (256 threads)
// ---------------------------------------------------------------------------
__device__ inline float blk_sum(float v, float* red, int tid) {
    red[tid] = v; __syncthreads();
    for (int s = 128; s > 0; s >>= 1) { if (tid < s) red[tid] += red[tid + s]; __syncthreads(); }
    float r = red[0]; __syncthreads(); return r;
}
__device__ inline float blk_max(float v, float* red, int tid) {
    red[tid] = v; __syncthreads();
    for (int s = 128; s > 0; s >>= 1) { if (tid < s) red[tid] = fmaxf(red[tid], red[tid + s]); __syncthreads(); }
    float r = red[0]; __syncthreads(); return r;
}

// ---------------------------------------------------------------------------
// K6: multi-factor attention per (b,h).  Algebra:
//   attn[f,l] = (qk*fw[l,f] + const)/8  with qk = (Wq y + bq) . wk
//   (const = Q.bk cancels in softmax over l)
//   correction[d] = wv[d]*S1 + F*bv[d],  S1 = sum_f sum_l softmax_l * fw
//   dots[b,h] = (Wsc correction + bsc) . y
// grid B*H, block 256
// ---------------------------------------------------------------------------
__global__ __launch_bounds__(256) void k_attn_pre(
    const float* __restrict__ yhat, const float* __restrict__ wh,
    const float* __restrict__ wf,
    const float* __restrict__ Wq, const float* __restrict__ bq,
    const float* __restrict__ wk, const float* __restrict__ wv,
    const float* __restrict__ bv,
    const float* __restrict__ Wsc, const float* __restrict__ bsc,
    float* __restrict__ dots)
{
    const int bh = blockIdx.x, b = bh / HORIZON, tid = threadIdx.x;
    const float inv_scale = 0.125f;              // 1/sqrt(D)

    __shared__ float y[DMODEL], corr[DMODEL], red[256];

    if (tid < DMODEL) y[tid] = yhat[(size_t)bh * DMODEL + tid];
    __syncthreads();

    // qk = sum_d (bq[d] + sum_e Wq[d,e] y[e]) * wk[d]
    float part = 0.f;
    if (tid < DMODEL) {
        float q = bq[tid];
        for (int e = 0; e < DMODEL; ++e) q += Wq[tid * DMODEL + e] * y[e];
        part = q * wk[tid];
    }
    const float qk = blk_sum(part, red, tid);    // uniform across block

    float S1 = 0.f;
    for (int f = 0; f < NFACT; ++f) {
        float m = -3.0e38f;
        for (int l = tid; l < LTOT; l += 256) {
            float fv = (l < T_LEN) ? wh[((size_t)(b * T_LEN + l)) * NFACT + f]
                                   : wf[((size_t)(b * HORIZON + (l - T_LEN))) * NFACT + f];
            m = fmaxf(m, qk * fv * inv_scale);
        }
        m = blk_max(m, red, tid);
        float se = 0.f, sf = 0.f;
        for (int l = tid; l < LTOT; l += 256) {
            float fv = (l < T_LEN) ? wh[((size_t)(b * T_LEN + l)) * NFACT + f]
                                   : wf[((size_t)(b * HORIZON + (l - T_LEN))) * NFACT + f];
            float ez = __expf(qk * fv * inv_scale - m);
            se += ez; sf += ez * fv;
        }
        se = blk_sum(se, red, tid);
        sf = blk_sum(sf, red, tid);
        S1 += sf / se;                           // uniform
    }

    if (tid < DMODEL) corr[tid] = wv[tid] * S1 + (float)NFACT * bv[tid];
    __syncthreads();

    float dp = 0.f;
    if (tid < DMODEL) {
        float c = bsc[tid];
        for (int e = 0; e < DMODEL; ++e) c += Wsc[tid * DMODEL + e] * corr[e];
        dp = c * y[tid];
    }
    float dot = blk_sum(dp, red, tid);
    if (tid == 0) dots[bh] = dot;
}

// ---------------------------------------------------------------------------
// K7: P = softmax_h(dots/8); final[d] = sum_h P[h]*y[b,h,d];
//     out[b,d,h] = y[b,h,d] + final[d]       (output layout [B,D,H])
// grid B, block 256
// ---------------------------------------------------------------------------
__global__ __launch_bounds__(256) void k_finalize(
    const float* __restrict__ yhat, const float* __restrict__ dots,
    float* __restrict__ out)
{
    const int b = blockIdx.x, tid = threadIdx.x;
    __shared__ float z[HORIZON], P[HORIZON];

    if (tid < HORIZON) z[tid] = dots[b * HORIZON + tid] * 0.125f;
    __syncthreads();
    if (tid == 0) {
        float m = -3.0e38f;
        for (int h = 0; h < HORIZON; ++h) m = fmaxf(m, z[h]);
        float s = 0.f;
        for (int h = 0; h < HORIZON; ++h) { P[h] = __expf(z[h] - m); s += P[h]; }
        float inv = 1.0f / s;
        for (int h = 0; h < HORIZON; ++h) P[h] *= inv;
    }
    __syncthreads();

    if (tid < DMODEL) {
        const int d = tid;
        float fin = 0.f;
        for (int h = 0; h < HORIZON; ++h)
            fin += P[h] * yhat[((size_t)(b * HORIZON + h)) * DMODEL + d];
        for (int h = 0; h < HORIZON; ++h)
            out[((size_t)(b * DMODEL + d)) * HORIZON + h] =
                yhat[((size_t)(b * HORIZON + h)) * DMODEL + d] + fin;
    }
}

// ---------------------------------------------------------------------------
// launch
// ---------------------------------------------------------------------------
extern "C" void kernel_launch(void* const* d_in, const int* in_sizes, int n_in,
                              void* d_out, int out_size, void* d_ws, size_t ws_size,
                              hipStream_t stream)
{
    const float* x   = (const float*)d_in[0];
    const float* wh  = (const float*)d_in[1];
    const float* wf  = (const float*)d_in[2];
    const float* Wi  = (const float*)d_in[3];
    const float* bi  = (const float*)d_in[4];
    const float* Wp1 = (const float*)d_in[5];
    const float* bp1 = (const float*)d_in[6];
    // d_in[7]/d_in[8] = Wp2/bp2: unused — reference discards block-2 conv output.
    const float* Wt1 = (const float*)d_in[9];
    const float* bt1 = (const float*)d_in[10];
    const float* Wt2 = (const float*)d_in[11];
    const float* bt2 = (const float*)d_in[12];
    const float* Wsd1= (const float*)d_in[13];
    const float* bsd1= (const float*)d_in[14];
    const float* Wsd2= (const float*)d_in[15];
    const float* bsd2= (const float*)d_in[16];
    const float* Wq  = (const float*)d_in[17];
    const float* bq  = (const float*)d_in[18];
    const float* wk  = (const float*)d_in[19];
    // d_in[20] = bk: unused — constant shift cancels in softmax over l.
    const float* wv  = (const float*)d_in[21];
    const float* bv  = (const float*)d_in[22];
    const float* Wsc = (const float*)d_in[23];
    const float* bsc = (const float*)d_in[24];

    const size_t BIG = (size_t)BATCH * DMODEL * T_LEN;   // 8388608 floats
    float* bufA = (float*)d_ws;         // xe, later tr2
    float* bufB = bufA + BIG;           // tr1
    float* bufC = bufB + BIG;           // seas1, later seas2
    float* bufD = bufC + BIG;           // u2 = s1 + tr1
    float* yhat = bufD + BIG;           // [B,H,D] = 98304
    float* dots = yhat + (size_t)BATCH * HORIZON * DMODEL;  // [B*H]
    // total ws use: 4*32MiB + ~0.4MiB

    // 1. xe = input projection
    k_input_proj<<<dim3(T_LEN / 256, BATCH), 256, 0, stream>>>(x, Wi, bi, bufA);

    // 2. block 1: tr1, seas1 from xe
    k_autoformer_core<<<BATCH * DMODEL, 256, 0, stream>>>(bufA, bufB, bufC);

    // 3. u2 = Wp1 @ (seas1 + tr1) + bp1 + tr1     (WMMA f16, LDS-staged B)
    k_conv_wmma<<<dim3(T_LEN / NCHUNK, BATCH), 256, 0, stream>>>(bufC, bufB, Wp1, bp1, bufD);

    // 4. block 2: tr2 (-> bufA), seas2 (-> bufC) from u2
    k_autoformer_core<<<BATCH * DMODEL, 256, 0, stream>>>(bufD, bufA, bufC);

    // 5. decoders on last timestep -> yhat [B,H,D]
    k_decoder<<<BATCH, 256, 0, stream>>>(bufA, bufC, Wt1, bt1, Wt2, bt2,
                                         Wsd1, bsd1, Wsd2, bsd2, yhat);

    // 6. per-(b,h) factor attention -> dots
    k_attn_pre<<<BATCH * HORIZON, 256, 0, stream>>>(yhat, wh, wf, Wq, bq, wk,
                                                    wv, bv, Wsc, bsc, dots);

    // 7. horizon softmax + residual -> out [B,D,H]
    k_finalize<<<BATCH, 256, 0, stream>>>(yhat, dots, (float*)d_out);
}
